// LDConv_DynamicSampling_70093866270953
// MI455X (gfx1250) — matrix operand order repaired and checked by reading, not compiled, not run on hardware
//
#include <hip/hip_runtime.h>

typedef __attribute__((ext_vector_type(16))) _Float16 v16h;
typedef __attribute__((ext_vector_type(8)))  float    v8f;

#define B_    4
#define C_    64
#define H_    128
#define W_    128
#define NP_   9
#define OUTC_ 128
#define HW_   (H_*W_)
#define HN_   (H_*NP_)            // 1152
#define M_    (B_*HN_*W_)         // 589824 GEMM rows
#define MT_   (M_/16)             // 36864 row tiles
#define EPS_  1e-5f

// workspace layout (bytes, 256-aligned)
#define WS_OFFS   0                         // 4*18*16384 f32   = 4,718,592 B
#define WS_APACK  4718592                   // M*64 f16         = 75,497,472 B
#define WS_BPACK  80216064                  // 8192 f16         = 16,384 B
#define WS_STATS  80232448                  // 256 f32          = 1,024 B

// ---------------- Kernel 1: 3x3 offset conv (64 -> 18 ch, SAME, zero pad) --------------
__global__ __launch_bounds__(256) void k_offset_conv(const float* __restrict__ x,
        const float* __restrict__ pw, const float* __restrict__ pb,
        float* __restrict__ offs) {
  __shared__ float xs[32][18][18];     // 41.5 KB: 32-channel slab of 16x16 tile + halo
  __shared__ float wsw[18][32][9];     // 20.7 KB
  const int b  = blockIdx.z;
  const int ty = threadIdx.y, tx = threadIdx.x;
  const int tid = ty*16 + tx;
  const int h0 = blockIdx.y*16, w0 = blockIdx.x*16;
  float acc[18];
  #pragma unroll
  for (int j = 0; j < 18; ++j) acc[j] = pb[j];
  for (int cp = 0; cp < 2; ++cp) {
    const int c0 = cp*32;
    __syncthreads();
    for (int idx = tid; idx < 32*18*18; idx += 256) {
      int ci = idx/324, rem = idx%324, ly = rem/18, lx = rem%18;
      int gy = h0+ly-1, gx = w0+lx-1;
      float v = 0.f;
      if ((unsigned)gy < (unsigned)H_ && (unsigned)gx < (unsigned)W_)
        v = x[((b*C_ + c0+ci)*H_ + gy)*W_ + gx];
      xs[ci][ly][lx] = v;
    }
    for (int idx = tid; idx < 18*32*9; idx += 256) {
      int j = idx/(32*9), rem = idx%(32*9), ci = rem/9, t = rem%9;
      wsw[j][ci][t] = pw[(j*C_ + c0+ci)*9 + t];
    }
    __syncthreads();
    for (int ci = 0; ci < 32; ++ci) {
      float xv[9];
      #pragma unroll
      for (int t = 0; t < 9; ++t) xv[t] = xs[ci][ty + t/3][tx + t%3];
      #pragma unroll
      for (int j = 0; j < 18; ++j) {
        float a = acc[j];
        #pragma unroll
        for (int t = 0; t < 9; ++t) a = fmaf(xv[t], wsw[j][ci][t], a);
        acc[j] = a;
      }
    }
  }
  const int h = h0+ty, w = w0+tx;
  #pragma unroll
  for (int j = 0; j < 18; ++j)
    offs[(b*18 + j)*HW_ + h*W_ + w] = acc[j];
}

// ------- Kernel 2: bilinear sampling, emit fp16 A-matrix in WMMA fragment layout -------
__global__ __launch_bounds__(256) void k_sample_pack(const float* __restrict__ x,
        const float* __restrict__ offs, _Float16* __restrict__ apack) {
  const int m  = blockIdx.x*256 + threadIdx.x;     // GEMM row: ((b*H + h)*NP + n)*W + w
  const int w  = m & (W_-1);
  const int hn = (m >> 7) % HN_;
  const int b  = m / (HN_*W_);
  const int h  = hn / NP_, n = hn % NP_;
  const float offx = offs[(b*18 + n      )*HW_ + h*W_ + w];
  const float offy = offs[(b*18 + NP_ + n)*HW_ + h*W_ + w];
  const float px = offx + (float)(n/3) + (float)h;   // p_n.x = n//3, p_0.x = h
  const float py = offy + (float)(n%3) + (float)w;   // p_n.y = n%3,  p_0.y = w
  const float ltx = floorf(px), lty = floorf(py);
  const float rbx = ltx + 1.f,  rby = lty + 1.f;
  const float gx0 = 1.f + (ltx - px), gy0 = 1.f + (lty - py);
  const float gx1 = 1.f - (rbx - px), gy1 = 1.f - (rby - py);
  const float g_lt = gx0*gy0, g_rb = gx1*gy1, g_lb = gx0*gy1, g_rt = gx1*gy0;
  auto clampi = [](float qx, float qy) -> int {
    int i = (int)(qx*(float)W_ + qy);               // matches reference: clip flat index
    return i < 0 ? 0 : (i > HW_-1 ? HW_-1 : i);
  };
  const int i_lt = clampi(ltx,lty), i_rb = clampi(rbx,rby);
  const int i_lb = clampi(ltx,rby), i_rt = clampi(rbx,lty);
  const int tile = m >> 4, r = m & 15;
  const float* xb = x + (size_t)b*C_*HW_;
  _Float16* base = apack + (size_t)tile*(2*32*16);
  #pragma unroll 4
  for (int c = 0; c < C_; ++c) {
    const float* xc = xb + c*HW_;
    float v = g_lt*xc[i_lt] + g_rb*xc[i_rb] + g_lb*xc[i_lb] + g_rt*xc[i_rt];
    // WMMA 16-bit A-fragment layout (16x32 per k-slice)
    const int ks = c >> 5, k = c & 31;
    const int lane = r + ((k>>3)&1)*16;
    const int hh   = (k&7) + ((k>>4)&1)*8;
    base[(ks*32 + lane)*16 + hh] = (_Float16)v;
  }
}

// ---------- Kernel 3: pack conv_w (128x64) into WMMA B-fragment layout (fp16) ----------
__global__ __launch_bounds__(512) void k_pack_w(const float* __restrict__ cw,
                                                _Float16* __restrict__ bpack) {
  const int t = threadIdx.x;            // 512 = 16 fragments * 32 lanes
  const int frag = t >> 5, lane = t & 31;
  const int nt = frag >> 1, ks = frag & 1;
  const int oc = nt*16 + (lane & 15);
  const int kbase = ks*32 + (lane >> 4)*8;
  #pragma unroll
  for (int hh = 0; hh < 16; ++hh)
    bpack[t*16 + hh] = (_Float16)cw[oc*C_ + kbase + (hh & 7) + (hh >> 3)*16];
}

// --------- Kernel 4: WMMA GEMM pass 1 — per-channel sum / sumsq for BatchNorm ----------
__global__ __launch_bounds__(256) void k_gemm_stats(const _Float16* __restrict__ apack,
        const _Float16* __restrict__ bpack,
        float* __restrict__ gsum, float* __restrict__ gsq) {
  __shared__ v16h  ldsB[512];                    // 16 KB weight fragments
  __shared__ float lsum[OUTC_], lsq[OUTC_];
  const int tid = threadIdx.x;
  if (tid < OUTC_) { lsum[tid] = 0.f; lsq[tid] = 0.f; }
  {
    const v16h* bp = (const v16h*)bpack;
    ldsB[tid]       = bp[tid];
    ldsB[tid + 256] = bp[tid + 256];
  }
  __syncthreads();
  const int lane = tid & 31;
  const int waveId = blockIdx.x*8 + (tid >> 5);
  const int tilesPerWave = MT_ / (gridDim.x*8);  // 36864 / 9216 = 4
  const v16h* ap = (const v16h*)apack;
  for (int i = 0; i < tilesPerWave; ++i) {
    const int tile = waveId*tilesPerWave + i;
    v16h a0 = ap[(tile*2 + 0)*32 + lane];
    v16h a1 = ap[(tile*2 + 1)*32 + lane];
    #pragma unroll
    for (int nt = 0; nt < 8; ++nt) {
      v16h b0 = ldsB[(nt*2 + 0)*32 + lane];
      v16h b1 = ldsB[(nt*2 + 1)*32 + lane];
      v8f acc = {};
      acc = __builtin_amdgcn_wmma_f32_16x16x32_f16(false, a0, false, b0, (short)0, acc, false, false);
      acc = __builtin_amdgcn_wmma_f32_16x16x32_f16(false, a1, false, b1, (short)0, acc, false, false);
      float s = 0.f, q = 0.f;
      #pragma unroll
      for (int v = 0; v < 8; ++v) { float f = acc[v]; s += f; q += f*f; }
      s += __shfl_xor(s, 16, 32);                // lanes l / l^16 share column l&15
      q += __shfl_xor(q, 16, 32);
      if (lane < 16) {
        atomicAdd(&lsum[nt*16 + lane], s);
        atomicAdd(&lsq [nt*16 + lane], q);
      }
    }
  }
  __syncthreads();
  if (tid < OUTC_) {
    atomicAdd(&gsum[tid], lsum[tid]);
    atomicAdd(&gsq [tid], lsq [tid]);
  }
}

// ----- Kernel 5: WMMA GEMM pass 2 — fused BatchNorm + SiLU, scatter to NCHW output -----
__global__ __launch_bounds__(256) void k_gemm_bn_silu(const _Float16* __restrict__ apack,
        const _Float16* __restrict__ bpack, const float* __restrict__ gsum,
        const float* __restrict__ gsq, const float* __restrict__ gamma,
        const float* __restrict__ beta, float* __restrict__ out) {
  __shared__ v16h ldsB[512];
  const int tid = threadIdx.x;
  {
    const v16h* bp = (const v16h*)bpack;
    ldsB[tid]       = bp[tid];
    ldsB[tid + 256] = bp[tid + 256];
  }
  __syncthreads();
  const int lane = tid & 31;
  const int waveId = blockIdx.x*8 + (tid >> 5);
  const int tilesPerWave = MT_ / (gridDim.x*8);
  const v16h* ap = (const v16h*)apack;
  const float invM = 1.f/(float)M_;
  float scl[8], sft[8];
  #pragma unroll
  for (int nt = 0; nt < 8; ++nt) {               // per-lane BN params (oc fixed per nt)
    const int oc = nt*16 + (lane & 15);
    const float mean = gsum[oc]*invM;
    const float var  = gsq[oc]*invM - mean*mean;
    const float sc   = gamma[oc]*rsqrtf(var + EPS_);
    scl[nt] = sc;
    sft[nt] = beta[oc] - mean*sc;
  }
  for (int i = 0; i < tilesPerWave; ++i) {
    const int tile = waveId*tilesPerWave + i;
    v16h a0 = ap[(tile*2 + 0)*32 + lane];
    v16h a1 = ap[(tile*2 + 1)*32 + lane];
    #pragma unroll
    for (int nt = 0; nt < 8; ++nt) {
      v16h b0 = ldsB[(nt*2 + 0)*32 + lane];
      v16h b1 = ldsB[(nt*2 + 1)*32 + lane];
      v8f acc = {};
      acc = __builtin_amdgcn_wmma_f32_16x16x32_f16(false, a0, false, b0, (short)0, acc, false, false);
      acc = __builtin_amdgcn_wmma_f32_16x16x32_f16(false, a1, false, b1, (short)0, acc, false, false);
      const int oc = nt*16 + (lane & 15);
      const int mbase = tile*16 + (lane >> 4)*8;
      #pragma unroll
      for (int v = 0; v < 8; ++v) {
        const int m  = mbase + v;
        const float y = acc[v]*scl[nt] + sft[nt];
        const float z = y / (1.f + __expf(-y));  // silu
        const int w  = m & (W_-1);
        const int hn = (m >> 7) % HN_;
        const int b  = m / (HN_*W_);
        out[(((size_t)b*OUTC_ + oc)*HN_ + hn)*W_ + w] = z;
      }
    }
  }
}

extern "C" void kernel_launch(void* const* d_in, const int* in_sizes, int n_in,
                              void* d_out, int out_size, void* d_ws, size_t ws_size,
                              hipStream_t stream) {
  const float* x     = (const float*)d_in[0];
  const float* pw    = (const float*)d_in[1];
  const float* pb    = (const float*)d_in[2];
  const float* cw    = (const float*)d_in[3];
  const float* gamma = (const float*)d_in[4];
  const float* beta  = (const float*)d_in[5];
  // d_in[6]=attn_w, d_in[7]=attn_b: 'density' is dead code in the reference — skipped.

  char* ws = (char*)d_ws;
  float*    offs  = (float*)   (ws + WS_OFFS);
  _Float16* apack = (_Float16*)(ws + WS_APACK);
  _Float16* bpack = (_Float16*)(ws + WS_BPACK);
  float*    stats = (float*)   (ws + WS_STATS);
  float* gsum = stats;
  float* gsq  = stats + OUTC_;

  hipMemsetAsync(stats, 0, 2*OUTC_*sizeof(float), stream);
  k_offset_conv <<<dim3(8,8,4),    dim3(16,16), 0, stream>>>(x, pw, pb, offs);
  k_sample_pack <<<dim3(M_/256),   dim3(256),   0, stream>>>(x, offs, apack);
  k_pack_w      <<<dim3(1),        dim3(512),   0, stream>>>(cw, bpack);
  k_gemm_stats  <<<dim3(1152),     dim3(256),   0, stream>>>(apack, bpack, gsum, gsq);
  k_gemm_bn_silu<<<dim3(1152),     dim3(256),   0, stream>>>(apack, bpack, gsum, gsq,
                                                             gamma, beta, (float*)d_out);
}
